// E_GAT_1778116460595
// MI455X (gfx1250) — matrix operand order repaired and checked by reading, not compiled, not run on hardware
//
#include <hip/hip_runtime.h>

// ---------------------------------------------------------------------------
// GAT layer for MI455X (gfx1250, wave32, WMMA).
// Heavy GEMM (message @ W, 8.6 GFLOP over a 134 MB stream) runs on
// v_wmma_f32_16x16x32_bf16 (f32 accumulate). Memory floor ~5.8 us @ 23.3 TB/s;
// bf16 WMMA keeps matrix-op count 8x below the f32 16x16x4 path so the kernel
// stays HBM-bound.
// ---------------------------------------------------------------------------

typedef __attribute__((ext_vector_type(16))) __bf16        v16bf;
typedef __attribute__((ext_vector_type(8)))  float         v8f;
typedef __attribute__((ext_vector_type(8)))  unsigned int  v8u;

#define AP 136   // msg-tile LDS pitch in ushorts (272 B rows, 16B aligned)
#define FP 132   // Hm LDS pitch in floats (dodges 64-bank conflicts on column reads)

__device__ __forceinline__ unsigned short f2bf(float f) {
    unsigned int u = __float_as_uint(f);
    u += 0x7FFFu + ((u >> 16) & 1u);          // round-to-nearest-even
    return (unsigned short)(u >> 16);
}
__device__ __forceinline__ unsigned int pk2(float a, float b) {
    return (unsigned int)f2bf(a) | ((unsigned int)f2bf(b) << 16);
}

// A-fragment (16x32 bf16, MxK) from LDS tile, per ISA 16-bit A layout:
// lanes 0-15: row=lane, K {k0..k0+7, k0+16..k0+23}; lanes 16-31: row=lane-16, +8.
__device__ __forceinline__ v16bf load_a_lds(const unsigned short* base, int mtile,
                                            int k0, int lane) {
    int m  = mtile * 16 + (lane & 15);
    int kb = k0 + ((lane >> 4) << 3);
    const unsigned short* p = base + m * AP + kb;
    uint4 lo = *(const uint4*)(p);        // K kb .. kb+7
    uint4 hi = *(const uint4*)(p + 16);   // K kb+16 .. kb+23
    v8u t = {lo.x, lo.y, lo.z, lo.w, hi.x, hi.y, hi.z, hi.w};
    return __builtin_bit_cast(v16bf, t);
}

// B-fragment (32x16 bf16, KxN) from W^T (row-major [n][k] bf16):
// lanes 0-15 hold K {k0..k0+7, k0+16..k0+23} of column n0+lane; lanes 16-31: +8.
__device__ __forceinline__ v16bf load_b_glb(const unsigned short* __restrict__ Wt,
                                            int n0, int k0, int lane) {
    int n  = n0 + (lane & 15);
    int kb = k0 + ((lane >> 4) << 3);
    const unsigned short* p = Wt + n * 128 + kb;
    uint4 lo = *(const uint4*)(p);
    uint4 hi = *(const uint4*)(p + 16);
    v8u t = {lo.x, lo.y, lo.z, lo.w, hi.x, hi.y, hi.z, hi.w};
    return __builtin_bit_cast(v16bf, t);
}

// Stage a 32x128 f32 tile -> bf16 LDS tile (256 threads, 16 elems each).
__device__ __forceinline__ void stage_tile(const float* __restrict__ src128,
                                           unsigned short* tile, int t) {
    int r = t >> 3, c0 = (t & 7) * 16;
    const float4* src = (const float4*)(src128 + (size_t)r * 128 + c0);
    float4 f0 = src[0], f1 = src[1], f2 = src[2], f3 = src[3];
    uint4* dst = (uint4*)(tile + r * AP + c0);
    dst[0] = make_uint4(pk2(f0.x, f0.y), pk2(f0.z, f0.w), pk2(f1.x, f1.y), pk2(f1.z, f1.w));
    dst[1] = make_uint4(pk2(f2.x, f2.y), pk2(f2.z, f2.w), pk2(f3.x, f3.y), pk2(f3.z, f3.w));
}

// ---------------------------------------------------------------------------
// Prep: W (DxH f32, row-major) -> W^T bf16 ([n][k]) in scratch.
// ---------------------------------------------------------------------------
__global__ void gat_prep_w(const float* __restrict__ W, unsigned short* __restrict__ Wt) {
    int idx = blockIdx.x * 256 + threadIdx.x;   // 64 blocks x 256 = 16384
    int n = idx >> 7, k = idx & 127;
    Wt[n * 128 + k] = f2bf(W[k * 128 + n]);
}

// ---------------------------------------------------------------------------
// hW = h @ W  (8192x128 @ 128x128), 32 rows per workgroup, WMMA bf16.
// ---------------------------------------------------------------------------
__global__ void gat_hw(const float* __restrict__ h, const unsigned short* __restrict__ Wt,
                       float* __restrict__ hW) {
    __shared__ __align__(16) unsigned short tile[32 * AP];
    int t  = threadIdx.x;
    int m0 = blockIdx.x * 32;
    stage_tile(h + (size_t)m0 * 128, tile, t);
    __syncthreads();

    int wave = t >> 5, lane = t & 31, n0 = wave * 16;
    v8f acc0 = {}, acc1 = {};
#pragma unroll
    for (int kk = 0; kk < 4; ++kk) {
        int k0 = kk * 32;
        v16bf bf = load_b_glb(Wt, n0, k0, lane);
        v16bf a0 = load_a_lds(tile, 0, k0, lane);
        v16bf a1 = load_a_lds(tile, 1, k0, lane);
        acc0 = __builtin_amdgcn_wmma_f32_16x16x32_bf16(false, a0, false, bf,
                                                       (short)0, acc0, false, false);
        acc1 = __builtin_amdgcn_wmma_f32_16x16x32_bf16(false, a1, false, bf,
                                                       (short)0, acc1, false, false);
    }
    // D layout: vgpr r -> row M=r (lanes 0-15) / M=8+r (lanes 16-31), col N = n0 + lane%16
    int col = n0 + (lane & 15);
    int rb  = m0 + ((lane >> 4) << 3);
#pragma unroll
    for (int r = 0; r < 8; ++r) hW[(size_t)(rb + r) * 128 + col] = acc0[r];
#pragma unroll
    for (int r = 0; r < 8; ++r) hW[(size_t)(rb + 16 + r) * 128 + col] = acc1[r];
}

// ---------------------------------------------------------------------------
// Main: one workgroup per (b,n). Hm tile via WMMA, then attention + output.
// ---------------------------------------------------------------------------
__global__ void gat_main(const float* __restrict__ msg, const int* __restrict__ mask,
                         const float* __restrict__ hW, const unsigned short* __restrict__ Wt,
                         const float* __restrict__ a_w, const float* __restrict__ out_w,
                         const float* __restrict__ out_b, float* __restrict__ out) {
    __shared__ __align__(16) unsigned short tile[32 * AP];  // message tile, bf16
    __shared__ __align__(16) float Hm[32 * FP];             // Hm tile, f32
    __shared__ __align__(16) float hWs[128];
    __shared__ __align__(16) float gRow[128];
    __shared__ float e2s[32];
    __shared__ float alphas[32];
    __shared__ float e1s;

    int t  = threadIdx.x;
    int bn = blockIdx.x;                                    // b*32 + n

    stage_tile(msg + (size_t)bn * 4096, tile, t);
    if (t < 128) hWs[t] = hW[(size_t)bn * 128 + t];
    __syncthreads();

    // --- Hm[a,:] = message[b,n,a,:] @ W --------------------------------------
    int wave = t >> 5, lane = t & 31, n0 = wave * 16;
    v8f acc0 = {}, acc1 = {};
#pragma unroll
    for (int kk = 0; kk < 4; ++kk) {
        int k0 = kk * 32;
        v16bf bf = load_b_glb(Wt, n0, k0, lane);
        v16bf a0 = load_a_lds(tile, 0, k0, lane);
        v16bf a1 = load_a_lds(tile, 1, k0, lane);
        acc0 = __builtin_amdgcn_wmma_f32_16x16x32_bf16(false, a0, false, bf,
                                                       (short)0, acc0, false, false);
        acc1 = __builtin_amdgcn_wmma_f32_16x16x32_bf16(false, a1, false, bf,
                                                       (short)0, acc1, false, false);
    }
    {
        int col = n0 + (lane & 15);
        int rb  = (lane >> 4) << 3;
#pragma unroll
        for (int r = 0; r < 8; ++r) Hm[(rb + r) * FP + col] = acc0[r];
#pragma unroll
        for (int r = 0; r < 8; ++r) Hm[(rb + 16 + r) * FP + col] = acc1[r];
    }
    __syncthreads();

    // --- e2[a] = Hm[a,:] . a2  (8 lanes per a, shuffle-reduce) ---------------
    {
        int a = t >> 3, j0 = (t & 7) * 16;
        float p = 0.f;
#pragma unroll
        for (int j = 0; j < 16; ++j) p += Hm[a * FP + j0 + j] * a_w[128 + j0 + j];
        p += __shfl_xor(p, 1);
        p += __shfl_xor(p, 2);
        p += __shfl_xor(p, 4);
        if ((t & 7) == 0) e2s[a] = p;
    }
    // --- e1 = hW[b,n,:] . a1 (wave 0) ----------------------------------------
    if (t < 32) {
        int j = t * 4;
        float p = hWs[j] * a_w[j] + hWs[j + 1] * a_w[j + 1] +
                  hWs[j + 2] * a_w[j + 2] + hWs[j + 3] * a_w[j + 3];
        p += __shfl_xor(p, 1);  p += __shfl_xor(p, 2);  p += __shfl_xor(p, 4);
        p += __shfl_xor(p, 8);  p += __shfl_xor(p, 16);
        if (t == 0) e1s = p;
    }
    __syncthreads();

    // --- masked softmax over a (reference: softmax(E*m), then alpha*m) -------
    if (t < 32) {
        float mv = (float)mask[(size_t)bn * 32 + t];
        float E  = e1s + e2s[t];
        E = (E >= 0.f) ? E : 0.2f * E;          // leaky relu (0.2)
        float Em = E * mv;                      // masked logits (0 where m==0)
        float mx = Em;
        mx = fmaxf(mx, __shfl_xor(mx, 16)); mx = fmaxf(mx, __shfl_xor(mx, 8));
        mx = fmaxf(mx, __shfl_xor(mx, 4));  mx = fmaxf(mx, __shfl_xor(mx, 2));
        mx = fmaxf(mx, __shfl_xor(mx, 1));
        float ex = __expf(Em - mx);
        float sm = ex;
        sm += __shfl_xor(sm, 16); sm += __shfl_xor(sm, 8); sm += __shfl_xor(sm, 4);
        sm += __shfl_xor(sm, 2);  sm += __shfl_xor(sm, 1);
        alphas[t] = (ex / sm) * mv;             // denominator includes masked exp(0)
    }
    __syncthreads();

    // --- g = hW + sum_a alpha[a] * Hm[a,:] -----------------------------------
    if (t < 128) {
        float wsum = 0.f;
#pragma unroll
        for (int a = 0; a < 32; ++a) wsum += alphas[a] * Hm[a * FP + t];
        gRow[t] = hWs[t] + wsum;
    }
    __syncthreads();

    // --- out = elu(g @ out_w^T + out_b) --------------------------------------
    if (t < 128) {
        float acc = out_b[t];
        const float4* g4 = (const float4*)gRow;
        const float4* w4 = (const float4*)(out_w + (size_t)t * 128);
#pragma unroll
        for (int jj = 0; jj < 32; ++jj) {
            float4 g = g4[jj], w = w4[jj];
            acc += g.x * w.x + g.y * w.y + g.z * w.z + g.w * w.w;
        }
        out[(size_t)bn * 128 + t] = (acc > 0.f) ? acc : (__expf(acc) - 1.f);
    }
}

// ---------------------------------------------------------------------------
extern "C" void kernel_launch(void* const* d_in, const int* in_sizes, int n_in,
                              void* d_out, int out_size, void* d_ws, size_t ws_size,
                              hipStream_t stream) {
    const float* h     = (const float*)d_in[0];   // (256,32,128)
    const float* msg   = (const float*)d_in[1];   // (256,32,32,128)
    const int*   mask  = (const int*)  d_in[2];   // (256,32,32)
    const float* W     = (const float*)d_in[3];   // (128,128)
    const float* a_w   = (const float*)d_in[4];   // (256,)
    const float* out_w = (const float*)d_in[5];   // (128,128)
    const float* out_b = (const float*)d_in[6];   // (128,)
    float*       out   = (float*)d_out;           // (256,32,128)

    // scratch layout: [0, 32KB) W^T bf16; [32KB, 32KB+4MB) hW f32
    unsigned short* Wt = (unsigned short*)d_ws;
    float*          hW = (float*)((char*)d_ws + 32768);

    gat_prep_w<<<64,   256, 0, stream>>>(W, Wt);
    gat_hw    <<<256,  256, 0, stream>>>(h, Wt, hW);
    gat_main  <<<8192, 256, 0, stream>>>(msg, mask, hW, Wt, a_w, out_w, out_b, out);
}
